// GraphSAGEModel_4844723109938
// MI455X (gfx1250) — matrix operand order repaired and checked by reading, not compiled, not run on hardware
//
#include <hip/hip_runtime.h>

typedef __attribute__((ext_vector_type(16))) _Float16 v16h;
typedef __attribute__((ext_vector_type(8)))  _Float16 v8h;
typedef __attribute__((ext_vector_type(8)))  float    v8f;

constexpr int N_NODES  = 50000;
constexpr int N_EDGES  = 800000;
constexpr int N_GRAPHS = 64;
constexpr int IN_CH    = 128;
constexpr int HID_CH   = 256;
constexpr int OUT_CH   = 256;
constexpr int HALF_OUT = 128;   // OUT_CH/2
constexpr int NCLS     = 10;

// ---------------- workspace layout (bytes) ----------------
constexpr size_t SZ_XF16  = (size_t)N_NODES * IN_CH * 2;      // 12.8 MB
constexpr size_t OFF_XF16 = 0;
constexpr size_t OFF_M1   = OFF_XF16 + SZ_XF16;               // mean1_f16, 12.8 MB
constexpr size_t SZ_M1    = (size_t)N_NODES * IN_CH * 2;
constexpr size_t OFF_M2   = 0;                                 // mean2_f16 aliases [x_f16, mean1_f16] (25.6 MB)
constexpr size_t OFF_AGG1 = OFF_M1 + SZ_M1;                    // 25.6 MB
constexpr size_t SZ_AGG1  = (size_t)N_NODES * IN_CH * 4;
constexpr size_t OFF_H1F16= OFF_AGG1;                          // h1_f16 aliases agg1 (both 25.6 MB)
constexpr size_t OFF_H1F32= OFF_AGG1 + SZ_AGG1;                // 51.2 MB
constexpr size_t SZ_H1F32 = (size_t)N_NODES * HID_CH * 4;
constexpr size_t OFF_AGG2 = OFF_H1F32 + SZ_H1F32;              // 51.2 MB
constexpr size_t SZ_AGG2  = (size_t)N_NODES * HID_CH * 4;
constexpr size_t OFF_H2   = OFF_AGG2;                          // h2_f32 aliases agg2
constexpr size_t OFF_CNT  = OFF_AGG2 + SZ_AGG2;
constexpr size_t SZ_CNT   = ((size_t)N_NODES * 4 + 255) & ~(size_t)255;
constexpr size_t OFF_W1LT = OFF_CNT + SZ_CNT;                  // [256][128] f16
constexpr size_t OFF_W1RT = OFF_W1LT + (size_t)HID_CH * IN_CH * 2;
constexpr size_t OFF_W2LT = OFF_W1RT + (size_t)HID_CH * IN_CH * 2;   // [256][256] f16
constexpr size_t OFF_W2RT = OFF_W2LT + (size_t)OUT_CH * HID_CH * 2;
constexpr size_t OFF_GSUM = OFF_W2RT + (size_t)OUT_CH * HID_CH * 2;  // [64][256] f32
constexpr size_t OFF_GCNT = OFF_GSUM + (size_t)N_GRAPHS * OUT_CH * 4;

// ---------------- utility kernels ----------------
__global__ void zero_f32(float* p, int n) {
    int i = blockIdx.x * blockDim.x + threadIdx.x;
    if (i < n) p[i] = 0.0f;
}

__global__ void f32_to_f16(const float* in, _Float16* out, int n) {
    int i = blockIdx.x * blockDim.x + threadIdx.x;
    if (i < n) out[i] = (_Float16)in[i];
}

// W: [K][N] fp32  ->  Wt: [N][K] f16 (fragment-friendly: contiguous K per column)
__global__ void w_transpose_f16(const float* W, _Float16* Wt, int K, int N) {
    int i = blockIdx.x * blockDim.x + threadIdx.x;
    if (i < K * N) {
        int n = i / K, k = i % K;
        Wt[i] = (_Float16)W[(size_t)k * N + n];
    }
}

__global__ void degree_kernel(const long long* dst, float* cnt, int E) {
    int e = blockIdx.x * blockDim.x + threadIdx.x;
    if (e < E) atomicAdd(&cnt[(int)dst[e]], 1.0f);
}

// per-edge scatter-add of C-channel features (4 channels per thread)
__global__ void scatter_add(const long long* src, const long long* dst,
                            const float* feat, float* agg, int E, int C) {
    int t = blockIdx.x * blockDim.x + threadIdx.x;
    int per_edge = C >> 2;
    if (t >= E * per_edge) return;
    int e  = t / per_edge;
    int c4 = (t % per_edge) << 2;
    int s = (int)src[e];
    int d = (int)dst[e];
    const float4 v = *(const float4*)(feat + (size_t)s * C + c4);
    float* base = agg + (size_t)d * C + c4;
    atomicAdd(base + 0, v.x);
    atomicAdd(base + 1, v.y);
    atomicAdd(base + 2, v.z);
    atomicAdd(base + 3, v.w);
}

__global__ void mean_to_f16(const float* agg, const float* cnt, _Float16* out, int N, int C) {
    int i = blockIdx.x * blockDim.x + threadIdx.x;
    if (i < N * C) {
        int node = i / C;
        out[i] = (_Float16)(agg[i] / fmaxf(cnt[node], 1.0f));
    }
}

// ---------------- WMMA SAGE GEMM ----------------
// out[row, 0:256] = relu( Aagg[row,:] @ Wl + Aself[row,:] @ Wr + bias )
// Wl/Wr pre-transposed as [256][K] f16.  Block = 16 rows, 8 waves x 2 col-tiles.
__device__ __forceinline__ v16h load_a_frag(const _Float16* A, int row, int K, int k0, int kh) {
    const v8h lo = *(const v8h*)(A + (size_t)row * K + k0 + kh * 16);
    const v8h hi = *(const v8h*)(A + (size_t)row * K + k0 + kh * 16 + 8);
    v16h a;
#pragma unroll
    for (int i = 0; i < 8; i++) { a[i] = lo[i]; a[i + 8] = hi[i]; }
    return a;
}

template <int K>
__global__ __launch_bounds__(256) void sage_gemm(
    const _Float16* __restrict__ Aagg, const _Float16* __restrict__ Aself,
    const _Float16* __restrict__ Wlt,  const _Float16* __restrict__ Wrt,
    const float* __restrict__ bias, float* __restrict__ out_f32,
    _Float16* __restrict__ out_f16) {

    const int lane  = threadIdx.x & 31;
    const int wave  = threadIdx.x >> 5;          // 0..7
    const int mbase = blockIdx.x * 16;
    const int m     = lane & 15;
    const int kh    = lane >> 4;                 // 0/1: K-half group
    const int row   = mbase + m;
    const int n0    = (wave * 2 + 0) * 16 + m;   // B/D column for this lane
    const int n1    = (wave * 2 + 1) * 16 + m;

    v8f acc0 = {};
    v8f acc1 = {};

#pragma unroll
    for (int k0 = 0; k0 < K; k0 += 32) {
        const v16h aA = load_a_frag(Aagg,  row, K, k0, kh);
        const v16h aS = load_a_frag(Aself, row, K, k0, kh);
        const int kb = k0 + kh * 16;
        const v16h bL0 = *(const v16h*)(Wlt + (size_t)n0 * K + kb);
        const v16h bR0 = *(const v16h*)(Wrt + (size_t)n0 * K + kb);
        const v16h bL1 = *(const v16h*)(Wlt + (size_t)n1 * K + kb);
        const v16h bR1 = *(const v16h*)(Wrt + (size_t)n1 * K + kb);
        acc0 = __builtin_amdgcn_wmma_f32_16x16x32_f16(false, aA, false, bL0, (short)0, acc0, false, false);
        acc0 = __builtin_amdgcn_wmma_f32_16x16x32_f16(false, aS, false, bR0, (short)0, acc0, false, false);
        acc1 = __builtin_amdgcn_wmma_f32_16x16x32_f16(false, aA, false, bL1, (short)0, acc1, false, false);
        acc1 = __builtin_amdgcn_wmma_f32_16x16x32_f16(false, aS, false, bR1, (short)0, acc1, false, false);
    }

    // D layout: lane holds column n, VGPR v holds row m = 8*(lane>>4)+v
    const float b0 = bias[n0];
    const float b1 = bias[n1];
#pragma unroll
    for (int v = 0; v < 8; v++) {
        const int r = mbase + 8 * kh + v;
        float v0 = fmaxf(acc0[v] + b0, 0.0f);
        float v1 = fmaxf(acc1[v] + b1, 0.0f);
        out_f32[(size_t)r * 256 + n0] = v0;
        out_f32[(size_t)r * 256 + n1] = v1;
        if (out_f16) {
            out_f16[(size_t)r * 256 + n0] = (_Float16)v0;
            out_f16[(size_t)r * 256 + n1] = (_Float16)v1;
        }
    }
}

// ---------------- pooling + head ----------------
__global__ void pool_sum(const float* h, const long long* batch, float* gsum, int N, int C) {
    int i = blockIdx.x * blockDim.x + threadIdx.x;
    if (i < N * C) {
        int node = i / C, c = i % C;
        atomicAdd(&gsum[(int)batch[node] * C + c], h[i]);
    }
}

__global__ void pool_cnt(const long long* batch, float* gcnt, int N) {
    int i = blockIdx.x * blockDim.x + threadIdx.x;
    if (i < N) atomicAdd(&gcnt[(int)batch[i]], 1.0f);
}

__global__ __launch_bounds__(256) void head_kernel(
    const float* __restrict__ gsum, const float* __restrict__ gcnt,
    const float* __restrict__ Wl1, const float* __restrict__ bl1,
    const float* __restrict__ Wl2, const float* __restrict__ bl2,
    const float* __restrict__ mask, float* __restrict__ out) {

    __shared__ float t[N_GRAPHS * HALF_OUT];   // 32 KB
    for (int i = threadIdx.x; i < N_GRAPHS * HALF_OUT; i += blockDim.x) {
        int g = i / HALF_OUT, j = i % HALF_OUT;
        float inv = 1.0f / fmaxf(gcnt[g], 1.0f);
        float s = bl1[j];
        for (int c = 0; c < OUT_CH; c++)
            s += (gsum[g * OUT_CH + c] * inv) * Wl1[c * HALF_OUT + j];
        t[i] = s * mask[g * HALF_OUT + j];
    }
    __syncthreads();
    for (int i = threadIdx.x; i < N_GRAPHS * NCLS; i += blockDim.x) {
        int g = i / NCLS, k = i % NCLS;
        float s = bl2[k];
        for (int j = 0; j < HALF_OUT; j++)
            s += t[g * HALF_OUT + j] * Wl2[j * NCLS + k];
        out[i] = s;
    }
}

// ---------------- launcher ----------------
static inline int blocks_for(long long n, int bs) { return (int)((n + bs - 1) / bs); }

extern "C" void kernel_launch(void* const* d_in, const int* in_sizes, int n_in,
                              void* d_out, int out_size, void* d_ws, size_t ws_size,
                              hipStream_t stream) {
    const float*     x    = (const float*)d_in[0];
    const long long* ei   = (const long long*)d_in[1];
    const long long* bat  = (const long long*)d_in[2];
    const float* W1l = (const float*)d_in[3];
    const float* b1l = (const float*)d_in[4];
    const float* W1r = (const float*)d_in[5];
    const float* W2l = (const float*)d_in[6];
    const float* b2l = (const float*)d_in[7];
    const float* W2r = (const float*)d_in[8];
    const float* Wl1 = (const float*)d_in[9];
    const float* bl1 = (const float*)d_in[10];
    const float* Wl2 = (const float*)d_in[11];
    const float* bl2 = (const float*)d_in[12];
    const float* msk = (const float*)d_in[13];

    const long long* src = ei;
    const long long* dst = ei + N_EDGES;

    char* ws = (char*)d_ws;
    _Float16* x_f16  = (_Float16*)(ws + OFF_XF16);
    _Float16* m1_f16 = (_Float16*)(ws + OFF_M1);
    _Float16* m2_f16 = (_Float16*)(ws + OFF_M2);
    float*    agg1   = (float*)(ws + OFF_AGG1);
    _Float16* h1_f16 = (_Float16*)(ws + OFF_H1F16);
    float*    h1_f32 = (float*)(ws + OFF_H1F32);
    float*    agg2   = (float*)(ws + OFF_AGG2);
    float*    h2_f32 = (float*)(ws + OFF_H2);
    float*    cnt    = (float*)(ws + OFF_CNT);
    _Float16* W1lt   = (_Float16*)(ws + OFF_W1LT);
    _Float16* W1rt   = (_Float16*)(ws + OFF_W1RT);
    _Float16* W2lt   = (_Float16*)(ws + OFF_W2LT);
    _Float16* W2rt   = (_Float16*)(ws + OFF_W2RT);
    float*    gsum   = (float*)(ws + OFF_GSUM);
    float*    gcnt   = (float*)(ws + OFF_GCNT);

    const int BS = 256;
    float* out = (float*)d_out;

    // 0) zero accumulators (fresh every call: ws is not re-poisoned between replays)
    zero_f32<<<blocks_for((long long)N_NODES * IN_CH, BS), BS, 0, stream>>>(agg1, N_NODES * IN_CH);
    zero_f32<<<blocks_for((long long)N_NODES * HID_CH, BS), BS, 0, stream>>>(agg2, N_NODES * HID_CH);
    zero_f32<<<blocks_for(N_NODES, BS), BS, 0, stream>>>(cnt, N_NODES);
    zero_f32<<<blocks_for(N_GRAPHS * OUT_CH, BS), BS, 0, stream>>>(gsum, N_GRAPHS * OUT_CH);
    zero_f32<<<1, 64, 0, stream>>>(gcnt, N_GRAPHS);

    // 1) precision conversion / weight transposition
    f32_to_f16<<<blocks_for((long long)N_NODES * IN_CH, BS), BS, 0, stream>>>(x, x_f16, N_NODES * IN_CH);
    w_transpose_f16<<<blocks_for(IN_CH * HID_CH, BS), BS, 0, stream>>>(W1l, W1lt, IN_CH, HID_CH);
    w_transpose_f16<<<blocks_for(IN_CH * HID_CH, BS), BS, 0, stream>>>(W1r, W1rt, IN_CH, HID_CH);
    w_transpose_f16<<<blocks_for(HID_CH * OUT_CH, BS), BS, 0, stream>>>(W2l, W2lt, HID_CH, OUT_CH);
    w_transpose_f16<<<blocks_for(HID_CH * OUT_CH, BS), BS, 0, stream>>>(W2r, W2rt, HID_CH, OUT_CH);

    // 2) degree (shared across both layers)
    degree_kernel<<<blocks_for(N_EDGES, BS), BS, 0, stream>>>(dst, cnt, N_EDGES);

    // 3) layer 1: scatter -> mean -> WMMA GEMM (+relu)
    scatter_add<<<blocks_for((long long)N_EDGES * (IN_CH / 4), BS), BS, 0, stream>>>(src, dst, x, agg1, N_EDGES, IN_CH);
    mean_to_f16<<<blocks_for((long long)N_NODES * IN_CH, BS), BS, 0, stream>>>(agg1, cnt, m1_f16, N_NODES, IN_CH);
    sage_gemm<IN_CH><<<N_NODES / 16, 256, 0, stream>>>(m1_f16, x_f16, W1lt, W1rt, b1l, h1_f32, h1_f16);

    // 4) layer 2
    scatter_add<<<blocks_for((long long)N_EDGES * (HID_CH / 4), BS), BS, 0, stream>>>(src, dst, h1_f32, agg2, N_EDGES, HID_CH);
    mean_to_f16<<<blocks_for((long long)N_NODES * HID_CH, BS), BS, 0, stream>>>(agg2, cnt, m2_f16, N_NODES, HID_CH);
    sage_gemm<HID_CH><<<N_NODES / 16, 256, 0, stream>>>(m2_f16, h1_f16, W2lt, W2rt, b2l, h2_f32, (_Float16*)nullptr);

    // 5) global mean pool + MLP head (dropout mask applied)
    pool_sum<<<blocks_for((long long)N_NODES * OUT_CH, BS), BS, 0, stream>>>(h2_f32, bat, gsum, N_NODES, OUT_CH);
    pool_cnt<<<blocks_for(N_NODES, BS), BS, 0, stream>>>(bat, gcnt, N_NODES);
    head_kernel<<<1, 256, 0, stream>>>(gsum, gcnt, Wl1, bl1, Wl2, bl2, msk, out);
}
